// MLP_50663434224314
// MI455X (gfx1250) — matrix operand order repaired and checked by reading, not compiled, not run on hardware
//
#include <hip/hip_runtime.h>

// ---------------- CDNA5 WMMA types ----------------
typedef __attribute__((ext_vector_type(16))) __bf16 v16bf;
typedef __attribute__((ext_vector_type(8)))  float  v8f;

#define DIM_    2048
#define HIDDEN_ 8192
#define GROUP_  64

// ---------- async global->LDS staging: 64B A-span + 64B B-span ----------
// IOFFSET applies to BOTH lds and global addresses (ISA 08_async_tensor §4.4),
// so one (lds,global) pair covers 4 chunks via offset:16/32/48.
__device__ __forceinline__ void stage_async64(unsigned int aoff, unsigned int boff,
                                              const __bf16* ga, const __bf16* gb) {
  asm volatile(
    "global_load_async_to_lds_b128 %0, %2, off\n\t"
    "global_load_async_to_lds_b128 %0, %2, off offset:16\n\t"
    "global_load_async_to_lds_b128 %0, %2, off offset:32\n\t"
    "global_load_async_to_lds_b128 %0, %2, off offset:48\n\t"
    "global_load_async_to_lds_b128 %1, %3, off\n\t"
    "global_load_async_to_lds_b128 %1, %3, off offset:16\n\t"
    "global_load_async_to_lds_b128 %1, %3, off offset:32\n\t"
    "global_load_async_to_lds_b128 %1, %3, off offset:48"
    :: "v"(aoff), "v"(boff), "v"(ga), "v"(gb) : "memory");
}
__device__ __forceinline__ void wait_asynccnt0() {
  asm volatile("s_wait_asynccnt 0x0" ::: "memory");
}

// ---------------- ternary weight quantization (bf16 out) ----------------
__global__ void quant_ternary_k(const float* __restrict__ w,
                                __bf16* __restrict__ wq, int ngroups) {
  int g = blockIdx.x * blockDim.x + threadIdx.x;
  if (g >= ngroups) return;
  const float4* p4 = (const float4*)(w + (size_t)g * GROUP_);
  float wb[GROUP_];
  float s = 0.f;
  #pragma unroll
  for (int i = 0; i < GROUP_ / 4; ++i) {
    float4 v = p4[i];
    float f0 = (float)(__bf16)v.x, f1 = (float)(__bf16)v.y;
    float f2 = (float)(__bf16)v.z, f3 = (float)(__bf16)v.w;
    wb[4*i+0] = f0; wb[4*i+1] = f1; wb[4*i+2] = f2; wb[4*i+3] = f3;
    s += fabsf(f0) + fabsf(f1) + fabsf(f2) + fabsf(f3);
  }
  s *= (1.0f / GROUP_);
  s = fmaxf(s, 1e-8f);
  float inv = 1.0f / s;
  uint4* o4 = (uint4*)(wq + (size_t)g * GROUP_);
  #pragma unroll
  for (int c = 0; c < GROUP_ / 8; ++c) {
    union { __bf16 b[8]; uint4 q; } u;
    #pragma unroll
    for (int j = 0; j < 8; ++j) {
      float q = rintf(wb[8*c+j] * inv);          // round half-to-even
      q = fminf(1.f, fmaxf(-1.f, q));
      u.b[j] = (__bf16)(q * s);
    }
    o4[c] = u.q;
  }
}

// ---------------- f32 -> bf16 cast (8 elems/thread) ----------------
__global__ void cast_bf16_k(const float* __restrict__ x,
                            __bf16* __restrict__ xb, int n) {
  int i = (blockIdx.x * blockDim.x + threadIdx.x) * 8;
  if (i >= n) return;
  const float4* p = (const float4*)(x + i);
  float4 a = p[0], b = p[1];
  union { __bf16 bb[8]; uint4 q; } u;
  u.bb[0] = (__bf16)a.x; u.bb[1] = (__bf16)a.y;
  u.bb[2] = (__bf16)a.z; u.bb[3] = (__bf16)a.w;
  u.bb[4] = (__bf16)b.x; u.bb[5] = (__bf16)b.y;
  u.bb[6] = (__bf16)b.z; u.bb[7] = (__bf16)b.w;
  *(uint4*)(xb + i) = u.q;
}

// ---------------- row sum-of-squares (for fused rms-norm) ----------------
__global__ void rowss_k(const __bf16* __restrict__ h,
                        float* __restrict__ ss, int K) {
  int m = blockIdx.x;
  const uint4* p = (const uint4*)(h + (size_t)m * K);   // 8 bf16 / uint4
  int nvec = K / 8;
  float s = 0.f;
  for (int i = threadIdx.x; i < nvec; i += blockDim.x) {
    union { uint4 q; __bf16 b[8]; } u;
    u.q = p[i];
    #pragma unroll
    for (int j = 0; j < 8; ++j) { float f = (float)u.b[j]; s += f * f; }
  }
  __shared__ float red[256];
  red[threadIdx.x] = s;
  __syncthreads();
  for (int off = 128; off > 0; off >>= 1) {
    if ((int)threadIdx.x < off) red[threadIdx.x] += red[threadIdx.x + off];
    __syncthreads();
  }
  if (threadIdx.x == 0) ss[m] = red[0];
}

// ---------------- WMMA GEMM: C[M,N] = A[M,K] * B[N,K]^T ----------------
// Double-buffered LDS, async global->LDS staging, BK=64 (16 WMMA / barrier).
// MODE 0: epilogue relu(x)^2 -> bf16 (GEMM1); MODE 1: * rsqrt(ss/K+eps) -> f32.
constexpr int BM  = 128;
constexpr int BN  = 128;
constexpr int BK  = 64;
constexpr int LDT = 80;   // padded LDS row stride (bf16): 160B, 16B-aligned

union FragBF { v16bf v; uint4 q[2]; };

template<int MODE>
__global__ __launch_bounds__(256)
void gemm_wmma_k(const __bf16* __restrict__ A,   // [M,K] row-major
                 const __bf16* __restrict__ B,   // [N,K] row-major
                 void* __restrict__ Cout,
                 const float* __restrict__ ss,   // [M], MODE 1 only
                 int M, int N, int K) {
  __shared__ __align__(16) __bf16 As[2][BM * LDT];
  __shared__ __align__(16) __bf16 Bs[2][BN * LDT];

  const int t    = threadIdx.x;
  const int lane = t & 31;          // wave32
  const int wid  = t >> 5;          // 8 waves
  const int wm   = wid >> 1;        // 0..3  -> 32-row strip
  const int wn   = wid & 1;         // 0..1  -> 64-col strip

  const int rowBase = blockIdx.y * BM;
  const int colBase = blockIdx.x * BN;

  // staging coords: 256 threads x 64B contiguous span per tile
  const int lr = t >> 1;            // 0..127
  const int lc = (t & 1) << 5;      // 0 or 32 (bf16 elems)

  const __bf16* gA = A + (size_t)(rowBase + lr) * K + lc;
  const __bf16* gB = B + (size_t)(colBase + lr) * K + lc;

  // LDS byte offsets (flat LDS aperture keeps offset in addr[31:0])
  const unsigned int coff = (unsigned int)((lr * LDT + lc) * 2);
  const unsigned int aOfs[2] = {
      (unsigned int)(size_t)&As[0][0] + coff,
      (unsigned int)(size_t)&As[1][0] + coff };
  const unsigned int bOfs[2] = {
      (unsigned int)(size_t)&Bs[0][0] + coff,
      (unsigned int)(size_t)&Bs[1][0] + coff };

  v8f acc[2][4];
  #pragma unroll
  for (int i = 0; i < 2; ++i)
    #pragma unroll
    for (int j = 0; j < 4; ++j)
      #pragma unroll
      for (int e = 0; e < 8; ++e)
        acc[i][j][e] = 0.f;

  const int arow  = lane & 15;
  const int ahalf = lane >> 4;

  auto compute = [&](const __bf16* Ac, const __bf16* Bc) {
    #pragma unroll
    for (int ks = 0; ks < 2; ++ks) {              // two K=32 sub-steps
      const int ko = ks * 32;
      FragBF fa[2], fb[4];
      #pragma unroll
      for (int mt = 0; mt < 2; ++mt) {
        const __bf16* pp = Ac + (wm * 32 + mt * 16 + arow) * LDT + ko;
        fa[mt].q[0] = *(const uint4*)(pp + ahalf * 16);
        fa[mt].q[1] = *(const uint4*)(pp + 8 + ahalf * 16);
      }
      #pragma unroll
      for (int nt = 0; nt < 4; ++nt) {
        const __bf16* pp = Bc + (wn * 64 + nt * 16 + arow) * LDT + ko + ahalf * 16;
        fb[nt].q[0] = *(const uint4*)(pp + 0);
        fb[nt].q[1] = *(const uint4*)(pp + 8);
      }
      #pragma unroll
      for (int mt = 0; mt < 2; ++mt)
        #pragma unroll
        for (int nt = 0; nt < 4; ++nt)
          acc[mt][nt] = __builtin_amdgcn_wmma_f32_16x16x32_bf16(
              false, fa[mt].v, false, fb[nt].v,
              (short)0, acc[mt][nt], false, false);
    }
  };

  const int nk = K / BK;

  // prologue: stage tile 0 into buffer 0
  stage_async64(aOfs[0], bOfs[0], gA, gB);
  gA += BK; gB += BK;
  wait_asynccnt0();
  __syncthreads();

  for (int kt = 0; kt < nk - 1; ++kt) {
    const int cur = kt & 1, nxt = cur ^ 1;
    stage_async64(aOfs[nxt], bOfs[nxt], gA, gB);  // overlaps WMMAs below
    gA += BK; gB += BK;
    __builtin_prefetch(gA, 0, 1);                 // global_prefetch_b8 kt+2
    __builtin_prefetch(gB, 0, 1);
    compute(As[cur], Bs[cur]);
    wait_asynccnt0();                             // nxt fully in LDS
    __syncthreads();                              // one barrier / 16 WMMA
  }
  compute(As[(nk - 1) & 1], Bs[(nk - 1) & 1]);

  // Epilogue. D layout: lane L, VGPR v -> row = (L>>4)*8 + v, col = L&15
  const int crow0 = rowBase + wm * 32;
  const int ccol  = colBase + wn * 64 + (lane & 15);
  const int rhalf = (lane >> 4) * 8;

  if (MODE == 0) {
    __bf16* H = (__bf16*)Cout;
    #pragma unroll
    for (int mt = 0; mt < 2; ++mt)
      #pragma unroll
      for (int v = 0; v < 8; ++v) {
        int r = crow0 + mt * 16 + rhalf + v;
        #pragma unroll
        for (int nt = 0; nt < 4; ++nt) {
          float f = acc[mt][nt][v];
          f = fmaxf(f, 0.f);
          f = f * f;                           // relu(h)^2
          H[(size_t)r * N + ccol + nt * 16] = (__bf16)f;
        }
      }
  } else {
    float* O = (float*)Cout;
    const float rk = 1.0f / (float)K;
    #pragma unroll
    for (int mt = 0; mt < 2; ++mt)
      #pragma unroll
      for (int v = 0; v < 8; ++v) {
        int r = crow0 + mt * 16 + rhalf + v;
        float sc = rsqrtf(ss[r] * rk + 1.1920929e-07f);  // fused rms-norm
        #pragma unroll
        for (int nt = 0; nt < 4; ++nt)
          O[(size_t)r * N + ccol + nt * 16] = acc[mt][nt][v] * sc;
      }
  }
}

// ---------------- launch ----------------
extern "C" void kernel_launch(void* const* d_in, const int* in_sizes, int n_in,
                              void* d_out, int out_size, void* d_ws, size_t ws_size,
                              hipStream_t stream) {
  const float* x   = (const float*)d_in[0];   // [4,4096,2048] f32
  const float* Wfc = (const float*)d_in[1];   // [8192,2048]   f32
  const float* Wpj = (const float*)d_in[2];   // [2048,8192]   f32
  const int M = in_sizes[0] / DIM_;           // 16384 tokens

  char* w = (char*)d_ws;
  __bf16* xb  = (__bf16*)w; w += (size_t)M * DIM_ * 2;
  __bf16* wq1 = (__bf16*)w; w += (size_t)HIDDEN_ * DIM_ * 2;
  __bf16* wq2 = (__bf16*)w; w += (size_t)DIM_ * HIDDEN_ * 2;
  __bf16* hb  = (__bf16*)w; w += (size_t)M * HIDDEN_ * 2;
  float*  ssq = (float*)w;

  const int ng1 = HIDDEN_ * DIM_ / GROUP_;
  quant_ternary_k<<<(ng1 + 255) / 256, 256, 0, stream>>>(Wfc, wq1, ng1);
  const int ng2 = DIM_ * HIDDEN_ / GROUP_;
  quant_ternary_k<<<(ng2 + 255) / 256, 256, 0, stream>>>(Wpj, wq2, ng2);

  const int nx = M * DIM_;
  cast_bf16_k<<<(nx / 8 + 255) / 256, 256, 0, stream>>>(x, xb, nx);

  dim3 g1(HIDDEN_ / BN, M / BM);
  gemm_wmma_k<0><<<g1, 256, 0, stream>>>(xb, wq1, (void*)hb, nullptr,
                                         M, HIDDEN_, DIM_);

  rowss_k<<<M, 256, 0, stream>>>(hb, ssq, HIDDEN_);

  dim3 g2(DIM_ / BN, M / BM);
  gemm_wmma_k<1><<<g2, 256, 0, stream>>>(hb, wq2, d_out, ssq,
                                         M, DIM_, HIDDEN_);
}
